// PrimMLPs_88596585382210
// MI455X (gfx1250) — matrix operand order repaired and checked by reading, not compiled, not run on hardware
//
#include <hip/hip_runtime.h>
#include <hip/hip_bf16.h>

typedef __attribute__((ext_vector_type(16))) _Float16 v16h;
typedef __attribute__((ext_vector_type(8)))  _Float16 v8h;
typedef __attribute__((ext_vector_type(8)))  float    v8f;

#define T_TOK    16384
#define N_NODES  512
#define F_DIM    1024
#define M_DIM    256
#define N_SEM    8
#define OUT_Y    14
#define OUT_COLS 17
#define EPSV     1e-8f
#define TILE_M   64            // tokens per block (4 row-tiles of 16)

// ---------------------------------------------------------------------------
// WMMA helpers (CDNA5 wave32, V_WMMA_F32_16X16X32_F16)
// ---------------------------------------------------------------------------
__device__ __forceinline__ v8f wmma16(v16h a, v16h b, v8f c) {
  return __builtin_amdgcn_wmma_f32_16x16x32_f16(
      false, a, false, b, (short)0, c, false, false);
}

// A fragment (16x32 f16) from row-major LDS [rows][ldk] halves, rows mbase..+15.
// Lanes 0-15: M=lane, a[0..7]=K kb..kb+7, a[8..15]=K kb+16..kb+23,
// kb = k0 (+8 for lanes 16-31).
__device__ __forceinline__ v16h frag_a_lds(const _Float16* base, int ldk,
                                           int mbase, int k0, int lane) {
  int m  = mbase + (lane & 15);
  int kb = k0 + ((lane & 16) >> 1);
  const _Float16* p = base + m * ldk + kb;
  v8h lo = *(const v8h*)p;
  v8h hi = *(const v8h*)(p + 16);
  v16h a;
#pragma unroll
  for (int i = 0; i < 8; ++i) { a[i] = lo[i]; a[i + 8] = hi[i]; }
  return a;
}

// B fragment (32x16 f16) from transposed f16 weights Wt[n][k], ld=ldk.
// Lanes 0-15: N=lane, b[0..7]=K kb..kb+7, b[8..15]=K kb+16..kb+23,
// kb = k0 (+8 for lanes 16-31).
__device__ __forceinline__ v16h frag_b_glb(const _Float16* wt, int ldk,
                                           int nbase, int k0, int lane) {
  int n  = nbase + (lane & 15);
  int kb = k0 + ((lane & 16) >> 1);
  const _Float16* p = wt + (size_t)n * ldk + kb;
  v8h lo = *(const v8h*)p;
  v8h hi = *(const v8h*)(p + 16);
  v16h b;
#pragma unroll
  for (int i = 0; i < 8; ++i) { b[i] = lo[i]; b[i + 8] = hi[i]; }
  return b;
}

// C/D layout: lane -> N=lane&15; VGPR r -> M = r + 8*(lane>=16).
__device__ __forceinline__ void store_d_f16(v8f c, _Float16* dst, int ldn,
                                            int mbase, int nbase,
                                            const float* bias, int nbias,
                                            int lane, bool relu) {
  int n    = nbase + (lane & 15);
  float bv = (n < nbias) ? bias[n] : 0.0f;
  int moff = mbase + ((lane >> 4) << 3);
#pragma unroll
  for (int r = 0; r < 8; ++r) {
    float v = c[r] + bv;
    if (relu) v = fmaxf(v, 0.0f);
    dst[(moff + r) * ldn + n] = (_Float16)v;
  }
}

__device__ __forceinline__ void store_d_f32(v8f c, float* dst, int ldn,
                                            int mbase, int nbase,
                                            const float* bias, int nbias,
                                            int lane) {
  int n    = nbase + (lane & 15);
  float bv = (n < nbias) ? bias[n] : 0.0f;
  int moff = mbase + ((lane >> 4) << 3);
#pragma unroll
  for (int r = 0; r < 8; ++r) dst[(moff + r) * ldn + n] = c[r] + bv;
}

// ---------------------------------------------------------------------------
// Dispatch: bucket tokens by expert
// ---------------------------------------------------------------------------
__global__ void zero_counts_k(int* counts) {
  if (threadIdx.x < N_SEM) counts[threadIdx.x] = 0;
}

__global__ void build_perm_k(const int* __restrict__ sem, int* counts,
                             int* perm) {
  int t = blockIdx.x * blockDim.x + threadIdx.x;
  if (t < T_TOK) {
    int e = sem[t];
    int p = atomicAdd(counts + e, 1);
    perm[e * T_TOK + p] = t;
  }
}

// ---------------------------------------------------------------------------
// Weight convert: f32 [E][K][Nin] -> f16 transposed [E][Npad][K], zero-padded
// ---------------------------------------------------------------------------
__global__ void cvt_w_k(const float* __restrict__ W, _Float16* Wt,
                        int E, int K, int Nin, int Npad) {
  int idx = blockIdx.x * blockDim.x + threadIdx.x;
  int total = E * Npad * K;
  if (idx >= total) return;
  int k = idx % K;
  int rest = idx / K;
  int n = rest % Npad;
  int e = rest / Npad;
  float v = (n < Nin) ? W[((size_t)e * K + k) * Nin + n] : 0.0f;
  Wt[idx] = (_Float16)v;
}

// ---------------------------------------------------------------------------
// Stage one 64x64 f32 chunk of gathered tokens -> f16 LDS buffer
// ---------------------------------------------------------------------------
__device__ __forceinline__ void stage_chunk(const float* __restrict__ nodes,
                                            const int* sTok, _Float16* dstbuf,
                                            int kc, int tid) {
  int gr = tid >> 2;                     // row 0..63
  int gc = (tid & 3) * 16;               // 16-float segment
  const float4* src = (const float4*)(nodes + (size_t)sTok[gr] * F_DIM + kc + gc);
  _Float16* dst = dstbuf + gr * 64 + gc;
#pragma unroll
  for (int i = 0; i < 4; ++i) {
    float4 v = src[i];
    dst[i * 4 + 0] = (_Float16)v.x;
    dst[i * 4 + 1] = (_Float16)v.y;
    dst[i * 4 + 2] = (_Float16)v.z;
    dst[i * 4 + 3] = (_Float16)v.w;
  }
}

// One 64-wide K-chunk of the FC1 GEMM: 16 WMMAs into 8 accumulators.
__device__ __forceinline__ void fc1_step(const _Float16* cur,
                                         const _Float16* __restrict__ wt,
                                         int nb, int kc, int lane,
                                         v8f acc[4][2]) {
#pragma unroll
  for (int k2 = 0; k2 < 64; k2 += 32) {
    v16h b0 = frag_b_glb(wt, F_DIM, nb,      kc + k2, lane);
    v16h b1 = frag_b_glb(wt, F_DIM, nb + 16, kc + k2, lane);
#pragma unroll
    for (int rt = 0; rt < 4; ++rt) {
      v16h a = frag_a_lds(cur, 64, rt * 16, k2, lane);
      acc[rt][0] = wmma16(a, b0, acc[rt][0]);
      acc[rt][1] = wmma16(a, b1, acc[rt][1]);
    }
  }
}

// ---------------------------------------------------------------------------
// FC1 pass: dstH[64][256] = relu(X[64][1024] @ Wt^T + bias).
// X streamed in 64x64 f16 chunks, ping-pong double-buffered with
// compile-time-distinct buffers (keeps LDS accesses on the DS path).
// 8 waves x (2 n-tiles x 4 row-tiles) = 8 accumulators each.
// ---------------------------------------------------------------------------
__device__ __forceinline__ void pass_fc1(const float* __restrict__ nodes,
                                         const int* sTok,
                                         _Float16* sX0, _Float16* sX1,
                                         const _Float16* __restrict__ wt,
                                         const float* __restrict__ bias,
                                         _Float16* dstH,
                                         int tid, int lane, int wave) {
  v8f acc[4][2] = {};
  const int nb = wave * 32;

  stage_chunk(nodes, sTok, sX0, 0, tid);
  __syncthreads();

  for (int kc = 0; kc < F_DIM; kc += 128) {
    stage_chunk(nodes, sTok, sX1, kc + 64, tid);    // prefetch odd chunk
    fc1_step(sX0, wt, nb, kc, lane, acc);           // compute even chunk
    __syncthreads();
    if (kc + 128 < F_DIM)
      stage_chunk(nodes, sTok, sX0, kc + 128, tid); // prefetch next even
    fc1_step(sX1, wt, nb, kc + 64, lane, acc);      // compute odd chunk
    __syncthreads();
  }
#pragma unroll
  for (int rt = 0; rt < 4; ++rt) {
    store_d_f16(acc[rt][0], dstH, M_DIM, rt * 16, nb,      bias, M_DIM, lane, true);
    store_d_f16(acc[rt][1], dstH, M_DIM, rt * 16, nb + 16, bias, M_DIM, lane, true);
  }
}

// ---------------------------------------------------------------------------
// Fused MoE MLP: one block = one expert x one 64-token tile.
// LDS overlay: sXc1 doubles as the h2 buffer (only live after FC1 passes).
// ---------------------------------------------------------------------------
__global__ __launch_bounds__(256)
void moe_fused_k(const float* __restrict__ nodes,
                 const int* __restrict__ lengths,
                 const int* __restrict__ counts,
                 const int* __restrict__ perm,
                 const _Float16* __restrict__ W1t, const float* __restrict__ b1,
                 const _Float16* __restrict__ W2t, const float* __restrict__ b2,
                 const _Float16* __restrict__ W3t, const float* __restrict__ b3,
                 const _Float16* __restrict__ W4t, const float* __restrict__ b4,
                 const _Float16* __restrict__ R1t, const float* __restrict__ rb1,
                 const _Float16* __restrict__ R2t, const float* __restrict__ rb2,
                 float* __restrict__ out) {
  const int e    = blockIdx.y;
  const int tile = blockIdx.x;
  const int cnt  = counts[e];
  if (tile * TILE_M >= cnt) return;

  const int tid  = threadIdx.x;
  const int lane = tid & 31;
  const int wave = tid >> 5;

  __shared__ _Float16 sXc0[TILE_M * 64];       //  8 KB  X chunk buf 0
  __shared__ _Float16 sXc1[TILE_M * 64];       //  8 KB  X chunk buf 1 / h2
  __shared__ _Float16 sHbuf[TILE_M * M_DIM];   // 32 KB  r1 then h1
  __shared__ _Float16 sH3[TILE_M * 32];        //  4 KB
  __shared__ float    sY[TILE_M * 16];         //  4 KB
  __shared__ float    sRot[TILE_M * 16];       //  4 KB
  __shared__ int      sTok[TILE_M];
  _Float16* sH2 = sXc1;                        // alias: live only after FC1

  if (tid < TILE_M) {
    int idx = tile * TILE_M + tid;
    if (idx > cnt - 1) idx = cnt - 1;          // clamp partial tile
    sTok[tid] = perm[e * T_TOK + idx];
  }
  __syncthreads();

  // ---- pass R: r1 = relu(X@R1+rb1) into sHbuf -----------------------------
  pass_fc1(nodes, sTok, sXc0, sXc1, R1t + (size_t)e * M_DIM * F_DIM,
           rb1 + e * M_DIM, sHbuf, tid, lane, wave);
  __syncthreads();

  // ---- rot = r1@R2+rb2 [waves 0-3, one row-tile each; others park] --------
  if (wave < 4) {
    const _Float16* r2 = R2t + (size_t)e * 16 * M_DIM;
    v8f c = {};
    for (int k = 0; k < M_DIM; k += 32)
      c = wmma16(frag_a_lds(sHbuf, M_DIM, wave * 16, k, lane),
                 frag_b_glb(r2, M_DIM, 0, k, lane), c);
    store_d_f32(c, sRot, 16, wave * 16, 0, rb2 + e * 3, 3, lane);
  }
  __syncthreads();                             // r1 dead; sHbuf reusable

  // ---- pass H: h1 = relu(X@W1+b1) into sHbuf ------------------------------
  pass_fc1(nodes, sTok, sXc0, sXc1, W1t + (size_t)e * M_DIM * F_DIM,
           b1 + e * M_DIM, sHbuf, tid, lane, wave);
  __syncthreads();

  // ---- layer 2: h2 = relu(h1@W2+b2); 16 jobs (4rt x 4nt), 2 per wave ------
  {
    const _Float16* w2 = W2t + (size_t)e * 64 * M_DIM;
#pragma unroll
    for (int i = 0; i < 2; ++i) {
      int job = wave * 2 + i;
      int rt = job >> 2, nt = job & 3;
      v8f c = {};
      for (int k = 0; k < M_DIM; k += 32)
        c = wmma16(frag_a_lds(sHbuf, M_DIM, rt * 16, k, lane),
                   frag_b_glb(w2, M_DIM, nt * 16, k, lane), c);
      store_d_f16(c, sH2, 64, rt * 16, nt * 16, b2 + e * 64, 64, lane, true);
    }
  }
  __syncthreads();

  // ---- layer 3: h3 = relu(h2@W3+b3); 8 jobs (4rt x 2nt), 1 per wave -------
  {
    const _Float16* w3 = W3t + (size_t)e * 32 * 64;
    int rt = wave >> 1, nt = wave & 1;
    v8f c = {};
#pragma unroll
    for (int k = 0; k < 64; k += 32)
      c = wmma16(frag_a_lds(sH2, 64, rt * 16, k, lane),
                 frag_b_glb(w3, 64, nt * 16, k, lane), c);
    store_d_f16(c, sH3, 32, rt * 16, nt * 16, b3 + e * 32, 32, lane, true);
  }
  __syncthreads();

  // ---- layer 4: y = h3@W4+b4 [waves 0-3, one row-tile each] ---------------
  if (wave < 4) {
    const _Float16* w4 = W4t + (size_t)e * 16 * 32;
    v8f c = wmma16(frag_a_lds(sH3, 32, wave * 16, 0, lane),
                   frag_b_glb(w4, 32, 0, 0, lane), (v8f){});
    store_d_f32(c, sY, 16, wave * 16, 0, b4 + e * OUT_Y, OUT_Y, lane);
  }
  __syncthreads();

  // ---- masked write: out[tok][0..13]=y, [14..16]=rot; EPS beyond length ---
  for (int i = tid; i < TILE_M * OUT_COLS; i += 256) {
    int r = i / OUT_COLS;
    int c = i % OUT_COLS;
    if (tile * TILE_M + r < cnt) {
      int tok = sTok[r];
      int bg  = tok >> 9;                      // N_NODES = 512
      int nn  = tok & (N_NODES - 1);
      float v;
      if (nn >= lengths[bg]) v = EPSV;
      else v = (c < OUT_Y) ? sY[r * 16 + c] : sRot[r * 16 + (c - OUT_Y)];
      out[(size_t)tok * OUT_COLS + c] = v;
    }
  }
}

// ---------------------------------------------------------------------------
extern "C" void kernel_launch(void* const* d_in, const int* in_sizes, int n_in,
                              void* d_out, int out_size, void* d_ws,
                              size_t ws_size, hipStream_t stream) {
  const float* nodes   = (const float*)d_in[0];
  const int*   sem     = (const int*)d_in[1];
  const int*   lengths = (const int*)d_in[2];
  const float* W1 = (const float*)d_in[3];  const float* b1  = (const float*)d_in[4];
  const float* W2 = (const float*)d_in[5];  const float* b2  = (const float*)d_in[6];
  const float* W3 = (const float*)d_in[7];  const float* b3  = (const float*)d_in[8];
  const float* W4 = (const float*)d_in[9];  const float* b4  = (const float*)d_in[10];
  const float* R1 = (const float*)d_in[11]; const float* rb1 = (const float*)d_in[12];
  const float* R2 = (const float*)d_in[13]; const float* rb2 = (const float*)d_in[14];
  float* out = (float*)d_out;

  // workspace layout (256-byte aligned sections)
  char* ws = (char*)d_ws;
  int* counts = (int*)ws;
  int* perm   = (int*)(ws + 256);
  size_t off = 256 + (size_t)N_SEM * T_TOK * sizeof(int);
  _Float16* W1t = (_Float16*)(ws + off); off += (size_t)N_SEM * M_DIM * F_DIM * 2;
  _Float16* R1t = (_Float16*)(ws + off); off += (size_t)N_SEM * M_DIM * F_DIM * 2;
  _Float16* W2t = (_Float16*)(ws + off); off += (size_t)N_SEM * 64 * M_DIM * 2;
  _Float16* W3t = (_Float16*)(ws + off); off += (size_t)N_SEM * 32 * 64 * 2;
  _Float16* W4t = (_Float16*)(ws + off); off += (size_t)N_SEM * 16 * 32 * 2;
  _Float16* R2t = (_Float16*)(ws + off); off += (size_t)N_SEM * 16 * M_DIM * 2;

  zero_counts_k<<<1, 32, 0, stream>>>(counts);
  build_perm_k<<<T_TOK / 256, 256, 0, stream>>>(sem, counts, perm);

  auto cvt = [&](const float* W, _Float16* Wt, int K, int Nin, int Npad) {
    int total = N_SEM * Npad * K;
    cvt_w_k<<<(total + 255) / 256, 256, 0, stream>>>(W, Wt, N_SEM, K, Nin, Npad);
  };
  cvt(W1, W1t, F_DIM, M_DIM, M_DIM);
  cvt(R1, R1t, F_DIM, M_DIM, M_DIM);
  cvt(W2, W2t, M_DIM, 64, 64);
  cvt(W3, W3t, 64, 32, 32);
  cvt(W4, W4t, 32, OUT_Y, 16);
  cvt(R2, R2t, M_DIM, 3, 16);

  dim3 grid(T_TOK / TILE_M, N_SEM);
  moe_fused_k<<<grid, 256, 0, stream>>>(nodes, lengths, counts, perm,
                                        W1t, b1, W2t, b2, W3t, b3, W4t, b4,
                                        R1t, rb1, R2t, rb2, out);
}
